// SimpleModel_41394894798941
// MI455X (gfx1250) — matrix operand order repaired and checked by reading, not compiled, not run on hardware
//
#include <hip/hip_runtime.h>
#include <math.h>

// ---------------------------------------------------------------------------
// GCN graph-match model for MI455X (gfx1250, wave32, WMMA)
//   per graph: deg -> 3x [scatter-add over edges, (agg*inv_deg)@W+b via WMMA]
//   then column-mean readout, tiny MLP head.
// ---------------------------------------------------------------------------

typedef __attribute__((ext_vector_type(16))) __bf16 v16bf;
typedef __attribute__((ext_vector_type(8)))  __bf16 v8bf;
typedef __attribute__((ext_vector_type(8)))  float  v8f;

// ---------------- utility kernels ----------------

__global__ void zero_f32_kernel(float* __restrict__ p, int n) {
    int i = blockIdx.x * blockDim.x + threadIdx.x;
    if (i < n) p[i] = 0.0f;
}

__global__ void zero_i32_kernel(int* __restrict__ p, int n) {
    int i = blockIdx.x * blockDim.x + threadIdx.x;
    if (i < n) p[i] = 0;
}

__global__ void deg_count_kernel(const int* __restrict__ dst, int* __restrict__ deg, int E) {
    int i = blockIdx.x * blockDim.x + threadIdx.x;
    if (i < E) atomicAdd(&deg[dst[i]], 1);
}

__global__ void inv_deg_kernel(const int* __restrict__ deg, float* __restrict__ inv, int n) {
    int i = blockIdx.x * blockDim.x + threadIdx.x;
    if (i < n) inv[i] = 1.0f / fmaxf((float)deg[i], 1.0f);
}

// ---------------- edge scatter: agg[dst] += h[src] ----------------
// one wave32 per edge; lane handles 4 contiguous floats (float4 load,
// 4 hardware global_atomic_add_f32 into the L2-resident agg matrix).
__global__ void __launch_bounds__(256)
scatter_edges_kernel(const float* __restrict__ h,
                     const int* __restrict__ src, const int* __restrict__ dst,
                     float* __restrict__ agg, int E) {
    int e    = (blockIdx.x * 256 + threadIdx.x) >> 5;
    int lane = threadIdx.x & 31;
    if (e >= E) return;
    int s = src[e];
    int d = dst[e];
    const float4 v = *(const float4*)(h + s * 128 + lane * 4);
    float* a = agg + d * 128 + lane * 4;
    unsafeAtomicAdd(a + 0, v.x);
    unsafeAtomicAdd(a + 1, v.y);
    unsafeAtomicAdd(a + 2, v.z);
    unsafeAtomicAdd(a + 3, v.w);
}

// ---------------- WMMA GEMM: out[r,:] = (agg[r,:] * inv_deg[r]) @ W + b ----
// 256 threads = 8 waves; wave w owns output cols [16w,16w+16).
// Each block processes 16 row-tiles (256 rows); B fragments loaded once.
__global__ void __launch_bounds__(256)
gcn_gemm_kernel(const float* __restrict__ agg, const float* __restrict__ inv_deg,
                const float* __restrict__ W,   const float* __restrict__ bias,
                float* __restrict__ out, int n) {
    __shared__ __bf16 As[16 * 128];          // 4 KB A tile (bf16, pre-scaled)

    const int tid  = threadIdx.x;
    const int lane = tid & 31;
    const int wave = tid >> 5;               // 0..7
    const int ncol = wave * 16 + (lane & 15);

    // ---- load B fragments (K=128 as 4 chunks of 32), bf16 convert on the fly
    // layout: lanes 0-15 hold K=kchunk+0..15 (2 per VGPR), lanes 16-31 K=+16..31
    v16bf bfrag[4];
    {
        const int kb = (lane >> 4) * 16;
        #pragma unroll
        for (int c = 0; c < 4; ++c) {
            #pragma unroll
            for (int j = 0; j < 8; ++j) {
                bfrag[c][2 * j]     = (__bf16)W[(c * 32 + kb + 2 * j    ) * 128 + ncol];
                bfrag[c][2 * j + 1] = (__bf16)W[(c * 32 + kb + 2 * j + 1) * 128 + ncol];
            }
        }
    }
    const float bv = bias[ncol];

    // per-thread A-tile load coords: 8 contiguous elems per thread
    const int lr = tid >> 4;                 // row 0..15
    const int lc = (tid & 15) * 8;           // col 0..120

    const int row0 = blockIdx.x * 256;
    for (int t = 0; t < 16; ++t) {
        const int rbase = row0 + t * 16;
        const bool full = (rbase + 16 <= n);

        // ---- cooperative A-tile load: 16x128, scaled by inv_deg, bf16
        if (full) {
            const int grow = rbase + lr;
            const float id = inv_deg[grow];
            const float4 p0 = *(const float4*)(agg + grow * 128 + lc);
            const float4 p1 = *(const float4*)(agg + grow * 128 + lc + 4);
            v8bf vv;
            vv[0] = (__bf16)(p0.x * id); vv[1] = (__bf16)(p0.y * id);
            vv[2] = (__bf16)(p0.z * id); vv[3] = (__bf16)(p0.w * id);
            vv[4] = (__bf16)(p1.x * id); vv[5] = (__bf16)(p1.y * id);
            vv[6] = (__bf16)(p1.z * id); vv[7] = (__bf16)(p1.w * id);
            *(v8bf*)(As + lr * 128 + lc) = vv;
        } else {
            for (int i = tid; i < 16 * 128; i += 256) {
                int r = i >> 7, c = i & 127;
                int grow = rbase + r;
                float v = 0.0f;
                if (grow < n) v = agg[grow * 128 + c] * inv_deg[grow];
                As[i] = (__bf16)v;
            }
        }
        __syncthreads();

        // ---- A fragments, contiguous k order (same as B):
        //  lanes 0-15: e0..15 -> K kb2+0..15 (kb2=0)
        //  lanes16-31: same with kb2=16
        const int arow = lane & 15;
        const int kb2  = (lane >> 4) * 16;
        v8f acc;
        #pragma unroll
        for (int i = 0; i < 8; ++i) acc[i] = bv;   // bias is per-column

        #pragma unroll
        for (int c = 0; c < 4; ++c) {
            v16bf af;
            #pragma unroll
            for (int e = 0; e < 8; ++e) {
                af[e]     = As[arow * 128 + c * 32 + kb2 + e];
                af[8 + e] = As[arow * 128 + c * 32 + kb2 + 8 + e];
            }
            acc = __builtin_amdgcn_wmma_f32_16x16x32_bf16(
                false, af, false, bfrag[c], (short)0, acc, false, false);
        }

        // ---- C/D layout: VGPR i -> row i + 8*(lane>=16), col = lane&15
        const int mo = (lane >> 4) * 8;
        if (full) {
            const int base = (rbase + mo) * 128 + ncol;
            #pragma unroll
            for (int i = 0; i < 8; ++i) out[base + i * 128] = acc[i];
        } else {
            #pragma unroll
            for (int i = 0; i < 8; ++i) {
                int grow = rbase + mo + i;
                if (grow < n) out[grow * 128 + ncol] = acc[i];
            }
        }
        __syncthreads();
    }
}

// ---------------- column sum over nodes (for mean readout) ----------------
__global__ void __launch_bounds__(128)
colsum_kernel(const float* __restrict__ h, float* __restrict__ cs, int n) {
    int t  = threadIdx.x;                    // column 0..127
    int r0 = blockIdx.x * 256;
    int r1 = min(r0 + 256, n);
    float acc = 0.0f;
    for (int r = r0; r < r1; ++r) acc += h[r * 128 + t];
    unsafeAtomicAdd(&cs[t], acc);
}

// ---------------- readout + match head (single workgroup) ----------------
__global__ void __launch_bounds__(128)
head_kernel(const float* __restrict__ cs_p, const float* __restrict__ cs_s,
            const float* __restrict__ Wr,  const float* __restrict__ br,
            const float* __restrict__ Wm1, const float* __restrict__ bm1,
            const float* __restrict__ Wm2, const float* __restrict__ bm2,
            float* __restrict__ out, float invN) {
    __shared__ float mp[128], ms[128], g[256], m1[128];
    int t = threadIdx.x;
    mp[t] = cs_p[t] * invN;
    ms[t] = cs_s[t] * invN;
    __syncthreads();

    float ap = br[t], as = br[t];
    for (int k = 0; k < 128; ++k) {
        float w = Wr[k * 128 + t];
        ap += mp[k] * w;
        as += ms[k] * w;
    }
    g[t]       = 1.0f / (1.0f + __expf(-ap));
    g[128 + t] = 1.0f / (1.0f + __expf(-as));
    __syncthreads();

    float a = bm1[t];
    for (int k = 0; k < 256; ++k) a += g[k] * Wm1[k * 128 + t];
    m1[t] = a * Wm2[t];
    __syncthreads();

    if (t == 0) {
        float d = bm2[0];
        for (int k = 0; k < 128; ++k) d += m1[k];
        out[0] = 1.0f / (1.0f + __expf(-d));
    }
}

// ---------------------------------------------------------------------------

extern "C" void kernel_launch(void* const* d_in, const int* in_sizes, int n_in,
                              void* d_out, int out_size, void* d_ws, size_t ws_size,
                              hipStream_t stream) {
    // inputs in setup_inputs() order
    const float* feat_p = (const float*)d_in[0];
    const int*   src_p  = (const int*)  d_in[1];
    const int*   dst_p  = (const int*)  d_in[2];
    const float* feat_s = (const float*)d_in[3];
    const int*   src_s  = (const int*)  d_in[4];
    const int*   dst_s  = (const int*)  d_in[5];
    const float* W0 = (const float*)d_in[6];   const float* b0 = (const float*)d_in[7];
    const float* W1 = (const float*)d_in[8];   const float* b1 = (const float*)d_in[9];
    const float* W2 = (const float*)d_in[10];  const float* b2 = (const float*)d_in[11];
    const float* Wr = (const float*)d_in[12];  const float* br = (const float*)d_in[13];
    const float* Wm1= (const float*)d_in[14];  const float* bm1= (const float*)d_in[15];
    const float* Wm2= (const float*)d_in[16];  const float* bm2= (const float*)d_in[17];

    const int N = in_sizes[0] / 128;
    const int E = in_sizes[1];

    // workspace layout (matrix rows are 512 B, so alignment is natural)
    char* w = (char*)d_ws;
    const size_t matB = (size_t)N * 128 * sizeof(float);
    float* agg   = (float*)(w);
    float* h     = (float*)(w + matB);
    float* inv   = (float*)(w + 2 * matB);
    int*   deg   = (int*)  (w + 2 * matB + (size_t)N * 4);
    float* cs_p  = (float*)(w + 2 * matB + (size_t)N * 8);
    float* cs_s  = cs_p + 128;

    const int gN256 = (N + 255) / 256;
    const int gE256 = (E + 255) / 256;
    const int gScat = (E + 7) / 8;            // 8 waves/block, 1 edge/wave
    const int gGemm = (N + 255) / 256;        // 256 rows per block
    const int gCol  = (N + 255) / 256;

    const float* Ws[3] = {W0, W1, W2};
    const float* Bs[3] = {b0, b1, b2};

    struct G { const float* feat; const int* src; const int* dst; float* cs; };
    G graphs[2] = {{feat_p, src_p, dst_p, cs_p}, {feat_s, src_s, dst_s, cs_s}};

    for (int gi = 0; gi < 2; ++gi) {
        const G& gr = graphs[gi];
        // degrees -> inv_deg
        zero_i32_kernel<<<gN256, 256, 0, stream>>>(deg, N);
        deg_count_kernel<<<gE256, 256, 0, stream>>>(gr.dst, deg, E);
        inv_deg_kernel<<<gN256, 256, 0, stream>>>(deg, inv, N);

        const float* hin = gr.feat;
        for (int l = 0; l < 3; ++l) {
            zero_f32_kernel<<<(N * 128 + 255) / 256, 256, 0, stream>>>(agg, N * 128);
            scatter_edges_kernel<<<gScat, 256, 0, stream>>>(hin, gr.src, gr.dst, agg, E);
            gcn_gemm_kernel<<<gGemm, 256, 0, stream>>>(agg, inv, Ws[l], Bs[l], h, N);
            hin = h;
        }
        zero_f32_kernel<<<1, 128, 0, stream>>>(gr.cs, 128);
        colsum_kernel<<<gCol, 128, 0, stream>>>(h, gr.cs, N);
    }

    head_kernel<<<1, 128, 0, stream>>>(cs_p, cs_s, Wr, br, Wm1, bm1, Wm2, bm2,
                                       (float*)d_out, 1.0f / (float)N);
}
